// NTM_24713241822228
// MI455X (gfx1250) — matrix operand order repaired and checked by reading, not compiled, not run on hardware
//
#include <hip/hip_runtime.h>
#include <hip/hip_bf16.h>

// ---------------- problem constants (match reference) ----------------
constexpr int B    = 128;
constexpr int T    = 64;
constexpr int IN   = 512;
constexpr int CTRL = 512;
constexpr int N    = 128;   // memory rows
constexpr int MM   = 128;   // memory width
constexpr int S    = 3;     // shift window
constexpr int R    = 4;     // read heads
constexpr int W    = 1;     // write heads
constexpr int L    = 3 + MM + S;        // 134
constexpr int LW   = L + 2 * MM;        // 390
constexpr int NOUT = R * L + W * LW;    // 926
constexpr int KH   = IN + R * MM;       // 1024 controller K dim
constexpr int NT1  = CTRL / 16;         // 32 n-tiles for Wc
constexpr int NT2P = 60;                // 60 n-tiles (926 padded to 960) for Wk

typedef __attribute__((ext_vector_type(16))) __bf16 v16bf;
typedef __attribute__((ext_vector_type(8)))  float  v8f;
typedef __attribute__((ext_vector_type(4)))  int    v4i;

__device__ __forceinline__ __bf16 tobf(float f) { return (__bf16)f; }

// gfx1250 async global->LDS copy path (guarded; direct-load fallback)
#if defined(__AMDGCN__) && __has_builtin(__builtin_amdgcn_global_load_async_to_lds_b128) && __has_builtin(__builtin_amdgcn_s_wait_asynccnt)
#define USE_ASYNC_LDS 1
#else
#define USE_ASYNC_LDS 0
#endif

// ======================================================================
// K0: initialize recurrent state. Mem[:, N/2, :] = 1, everything else 0.
// ======================================================================
__global__ void k_init(float* __restrict__ rv, float* __restrict__ rw,
                       float* __restrict__ ww, float* __restrict__ Mem) {
  const int stride = gridDim.x * blockDim.x;
  const int tid0   = blockIdx.x * blockDim.x + threadIdx.x;
  for (int i = tid0; i < B * R * MM; i += stride) rv[i] = 0.f;
  for (int i = tid0; i < B * R * N;  i += stride) rw[i] = 0.f;
  for (int i = tid0; i < B * W * N;  i += stride) ww[i] = 0.f;
  for (int i = tid0; i < B * N * MM; i += stride) {
    int n = (i >> 7) & (N - 1);            // (i / MM) % N, MM = 128
    Mem[i] = (n == N / 2) ? 1.f : 0.f;
  }
}

// ======================================================================
// K-pack: convert a fp32 weight matrix (K x Nc, row-major) to bf16 in the
// exact WMMA 16-bit B-fragment lane order: one 32B fragment per (tile,lane).
// Lane l of tile (ktile,ntile) holds K = ktile*32 + (l>>4)*8 + (i&7) + (i>>3)*16,
// n = ntile*16 + (l&15).  One wave packs one tile.
// ======================================================================
__global__ void __launch_bounds__(256)
k_pack(const float* __restrict__ Wsrc, __bf16* __restrict__ pack,
       int K, int Nc, int NT) {
  const int tile = blockIdx.x * 8 + (threadIdx.x >> 5);
  const int lane = threadIdx.x & 31;
  const int ktiles = K / 32;
  if (tile >= ktiles * NT) return;
  const int kb = (tile / NT) * 32 + (lane >> 4) * 8;
  const int n  = (tile % NT) * 16 + (lane & 15);
  __bf16* dst = pack + ((size_t)tile * 32 + lane) * 16;
#pragma unroll
  for (int i = 0; i < 16; ++i) {
    const float v = (n < Nc) ? Wsrc[(size_t)(kb + (i & 7) + (i >> 3) * 16) * Nc + n] : 0.f;
    dst[i] = tobf(v);
  }
}

// ---------------- shared GEMM helper: build A frag from LDS tile ----------
__device__ __forceinline__ v16bf a_from_lds(const float* s_a, int lane) {
  const int r  = lane & 15;
  const int ka = (lane >> 4) * 8;
  v16bf a;
#pragma unroll
  for (int i = 0; i < 8; ++i) {
    a[i]     = tobf(s_a[r * 32 + ka + i]);
    a[8 + i] = tobf(s_a[r * 32 + ka + 16 + i]);
  }
  return a;
}

#if USE_ASYNC_LDS
__device__ __forceinline__ void async_stage16x32(const float* src_row_base,
                                                 size_t row_stride,
                                                 float* s_a, int tid) {
  // 128 threads each copy one 16B chunk: row r = tid>>3, floats c = (tid&7)*4
  const int r = tid >> 3, c = (tid & 7) * 4;
  const float* src = src_row_base + (size_t)r * row_stride + c;
  __builtin_amdgcn_global_load_async_to_lds_b128(
      (__attribute__((address_space(1))) v4i*)src,
      (__attribute__((address_space(3))) v4i*)&s_a[r * 32 + c], 0, 0);
}
#endif

// ======================================================================
// K1: out[b, t, :] = tanh([x_t | rv] @ Wc + bc)
// WMMA bf16 16x16x32, fp32 accumulate. One 16x16 tile per wave.
// grid = (B/16, CTRL/64), block = 128 (4 waves)
// ======================================================================
__global__ void __launch_bounds__(128)
k_controller(const float* __restrict__ x, const __bf16* __restrict__ packWc,
             const float* __restrict__ bc, const float* __restrict__ rv,
             float* __restrict__ outseq, int t) {
  const int tid   = threadIdx.x;
  const int wave  = tid >> 5;
  const int lane  = tid & 31;
  const int m0    = blockIdx.x * 16;                 // batch tile base
  const int ntile = blockIdx.y * 4 + wave;           // output col tile
  const int ncol  = ntile * 16 + (lane & 15);

  const float* __restrict__ xrow  = x  + (size_t)(m0 + (lane & 15)) * (T * IN) + (size_t)t * IN;
  const float* __restrict__ rvrow = rv + (size_t)(m0 + (lane & 15)) * (R * MM);
  const v16bf* __restrict__ bp    = (const v16bf*)packWc;

#if USE_ASYNC_LDS
  __shared__ float s_a[16 * 32];
#endif
  v8f acc = {};
  for (int kt = 0; kt < KH; kt += 32) {
    // uniform source select for this K-slab: x part or rv part
#if USE_ASYNC_LDS
    if (kt) __syncthreads();                         // WAR on s_a
    if (kt < IN)
      async_stage16x32(x + (size_t)m0 * (T * IN) + (size_t)t * IN + kt,
                       (size_t)(T * IN), s_a, tid);
    else
      async_stage16x32(rv + (size_t)m0 * (R * MM) + (kt - IN),
                       (size_t)(R * MM), s_a, tid);
    __builtin_amdgcn_s_wait_asynccnt(0);
    __syncthreads();                                 // RAW on s_a
    const v16bf a = a_from_lds(s_a, lane);
#else
    const float* rb = (kt < IN) ? (xrow + kt) : (rvrow + (kt - IN));
    const int ka = (lane >> 4) * 8;
    v16bf a;
#pragma unroll
    for (int i = 0; i < 8; ++i) {
      a[i]     = tobf(rb[ka + i]);
      a[8 + i] = tobf(rb[ka + 16 + i]);
    }
#endif
    if (lane == 0 && kt + 32 < KH)                   // stream next B slab
      __builtin_prefetch(&bp[((size_t)((kt >> 5) + 1) * NT1 + ntile) * 32], 0, 1);
    const v16bf bfr = bp[((size_t)(kt >> 5) * NT1 + ntile) * 32 + lane];
    acc = __builtin_amdgcn_wmma_f32_16x16x32_bf16(false, a, false, bfr,
                                                  (short)0, acc, false, false);
  }
  // D layout: VGPR v holds row m0 + (lane>>4)*8 + v, col = ncol
#pragma unroll
  for (int v = 0; v < 8; ++v) {
    const int m = m0 + (lane >> 4) * 8 + v;
    outseq[(size_t)m * (T * CTRL) + (size_t)t * CTRL + ncol] =
        tanhf(acc[v] + bc[ncol]);
  }
}

// ======================================================================
// K2: instrs = out_t @ Wk + bk   (M=128, K=512, N=926 padded to 960)
// grid = (B/16, 15), block = 128 (4 waves)
// ======================================================================
__global__ void __launch_bounds__(128)
k_instr(const float* __restrict__ outseq, const __bf16* __restrict__ packWk,
        const float* __restrict__ bk, float* __restrict__ instrs, int t) {
  const int tid   = threadIdx.x;
  const int wave  = tid >> 5;
  const int lane  = tid & 31;
  const int m0    = blockIdx.x * 16;
  const int ntile = blockIdx.y * 4 + wave;           // 0..59, pack is zero-padded
  const int ncol  = ntile * 16 + (lane & 15);

  const float* __restrict__ arow = outseq + (size_t)(m0 + (lane & 15)) * (T * CTRL) + (size_t)t * CTRL;
  const v16bf* __restrict__ bp   = (const v16bf*)packWk;

#if USE_ASYNC_LDS
  __shared__ float s_a[16 * 32];
#endif
  v8f acc = {};
  for (int kt = 0; kt < CTRL; kt += 32) {
#if USE_ASYNC_LDS
    if (kt) __syncthreads();
    async_stage16x32(outseq + (size_t)m0 * (T * CTRL) + (size_t)t * CTRL + kt,
                     (size_t)(T * CTRL), s_a, tid);
    __builtin_amdgcn_s_wait_asynccnt(0);
    __syncthreads();
    const v16bf a = a_from_lds(s_a, lane);
#else
    const int ka = (lane >> 4) * 8;
    v16bf a;
#pragma unroll
    for (int i = 0; i < 8; ++i) {
      a[i]     = tobf(arow[kt + ka + i]);
      a[8 + i] = tobf(arow[kt + ka + 16 + i]);
    }
#endif
    if (lane == 0 && kt + 32 < CTRL)
      __builtin_prefetch(&bp[((size_t)((kt >> 5) + 1) * NT2P + ntile) * 32], 0, 1);
    const v16bf bfr = bp[((size_t)(kt >> 5) * NT2P + ntile) * 32 + lane];
    acc = __builtin_amdgcn_wmma_f32_16x16x32_bf16(false, a, false, bfr,
                                                  (short)0, acc, false, false);
  }
  if (ncol < NOUT) {
#pragma unroll
    for (int v = 0; v < 8; ++v) {
      const int m = m0 + (lane >> 4) * 8 + v;
      instrs[(size_t)m * NOUT + ncol] = acc[v] + bk[ncol];
    }
  }
}

// ======================================================================
// K3: per-batch addressing + memory update + readout.
// One 256-thread block per batch; per-batch memory matrix is L2-resident.
// ======================================================================
__global__ void __launch_bounds__(256)
k_update(const float* __restrict__ instrs, float* __restrict__ Mem,
         float* __restrict__ rv, float* __restrict__ rw, float* __restrict__ ww) {
  const int b   = blockIdx.x;
  const int tid = threadIdx.x;
  const int wv  = tid >> 5, ln = tid & 31;

  __shared__ float s_inorm[N];       // 1/||Mem row||
  __shared__ float s_k[MM];          // current head key
  __shared__ float s_sim[N];         // sim / exp / sharpened (reused)
  __shared__ float s_wi[N + 2];      // circularly padded interpolated weight
  __shared__ float s_w[R + W][N];    // final weights per head
  __shared__ float s_par[8];         // beta,g,s0,s1,s2,t, inorm_k, red
  __shared__ float s_ea[2 * MM];     // erase | add vectors

  float*       __restrict__ Memb = Mem    + (size_t)b * N * MM;
  const float* __restrict__ ins  = instrs + (size_t)b * NOUT;

  // ---- 1) per-row inverse norms (wave-cooperative, coalesced) ----
  for (int n = wv; n < N; n += 8) {
    const float* row = Memb + n * MM;
    float s2 = 0.f;
    for (int j = ln; j < MM; j += 32) { float v = row[j]; s2 += v * v; }
    for (int off = 16; off; off >>= 1) s2 += __shfl_xor(s2, off, 32);
    if (ln == 0) s_inorm[n] = rsqrtf(fmaxf(s2, 1e-12f));
  }
  __syncthreads();

  // ---- 2) addressing for R read heads + W write head ----
  for (int h = 0; h < R + W; ++h) {
    const float* head = (h < R) ? (ins + h * L) : (ins + R * L + (h - R) * LW);
    float* wold = (h < R) ? (rw + ((size_t)b * R + h) * N) : (ww + (size_t)b * N);

    if (tid < MM) s_k[tid] = head[tid];
    if (tid == 0) {
      const float beta = expf(head[MM]);
      const float g    = 1.f / (1.f + expf(-head[MM + 1]));
      const float mx = fmaxf(head[MM + 2], fmaxf(head[MM + 3], head[MM + 4]));
      const float e0 = expf(head[MM + 2] - mx);
      const float e1 = expf(head[MM + 3] - mx);
      const float e2 = expf(head[MM + 4] - mx);
      const float es = e0 + e1 + e2;
      const float ti = head[MM + 2 + S];
      const float tp = ((ti > 20.f) ? ti : log1pf(expf(ti))) + 1.f;  // softplus+1
      float ks = 0.f;
      for (int m = 0; m < MM; ++m) { float v = head[m]; ks += v * v; }
      s_par[0] = beta; s_par[1] = g;
      s_par[2] = e0 / es; s_par[3] = e1 / es; s_par[4] = e2 / es;
      s_par[5] = tp; s_par[6] = rsqrtf(fmaxf(ks, 1e-12f));
    }
    __syncthreads();

    // cosine sim * beta (wave-cooperative, coalesced)
    for (int n = wv; n < N; n += 8) {
      const float* row = Memb + n * MM;
      float d = 0.f;
      for (int j = ln; j < MM; j += 32) d += row[j] * s_k[j];
      for (int off = 16; off; off >>= 1) d += __shfl_xor(d, off, 32);
      if (ln == 0) s_sim[n] = d * s_inorm[n] * s_par[6] * s_par[0];
    }
    __syncthreads();
    if (tid == 0) {                             // softmax over N=128
      float mx = -1e30f;
      for (int n = 0; n < N; ++n) mx = fmaxf(mx, s_sim[n]);
      float sm = 0.f;
      for (int n = 0; n < N; ++n) { float e = expf(s_sim[n] - mx); s_sim[n] = e; sm += e; }
      s_par[7] = 1.f / sm;
    }
    __syncthreads();
    if (tid < N) {                              // gate with previous weight
      const float g  = s_par[1];
      s_wi[tid + 1] = s_sim[tid] * s_par[7] * g + wold[tid] * (1.f - g);
    }
    __syncthreads();
    if (tid == 0) { s_wi[0] = s_wi[N]; s_wi[N + 1] = s_wi[1]; }  // circular pad
    __syncthreads();
    if (tid < N) {                              // conv + sharpen
      const float ws = s_wi[tid]     * s_par[2] +
                       s_wi[tid + 1] * s_par[3] +
                       s_wi[tid + 2] * s_par[4];
      s_sim[tid] = powf(fmaxf(ws, 0.f), s_par[5]);
    }
    __syncthreads();
    if (tid == 0) {
      float sm = 0.f;
      for (int n = 0; n < N; ++n) sm += s_sim[n];
      s_par[7] = 1.f / (sm + 1e-12f);
    }
    __syncthreads();
    if (tid < N) {                              // normalize, persist
      const float wn = s_sim[tid] * s_par[7];
      s_w[h][tid] = wn;
      wold[tid]   = wn;
    }
    __syncthreads();
  }

  // ---- 3) erase/add memory update (single write head) ----
  if (tid < MM) {
    s_ea[tid]      = ins[R * L + L + tid];        // erase
    s_ea[MM + tid] = ins[R * L + L + MM + tid];   // add
  }
  __syncthreads();
  for (int idx = tid; idx < N * MM; idx += 256) {
    const int n = idx >> 7, m = idx & (MM - 1);
    const float wn = s_w[R][n];
    const float v  = Memb[idx];
    Memb[idx] = v * (1.f - wn * s_ea[m]) + wn * s_ea[MM + m];
  }
  __threadfence_block();
  __syncthreads();

  // ---- 4) readout rv[r][m] = sum_n Mem[n][m] * w_r[n] ----
  {
    const int r0 = tid >> 7;          // 0..1
    const int m  = tid & (MM - 1);
    for (int r = r0; r < R; r += 2) {
      float acc = 0.f;
      for (int n = 0; n < N; ++n) acc += Memb[n * MM + m] * s_w[r][n];
      rv[((size_t)b * R + r) * MM + m] = acc;
    }
  }
}

// ======================================================================
extern "C" void kernel_launch(void* const* d_in, const int* in_sizes, int n_in,
                              void* d_out, int out_size, void* d_ws, size_t ws_size,
                              hipStream_t stream) {
  const float* x  = (const float*)d_in[0];
  const float* Wc = (const float*)d_in[1];
  const float* bc = (const float*)d_in[2];
  const float* Wk = (const float*)d_in[3];
  const float* bk = (const float*)d_in[4];
  float* out = (float*)d_out;

  float* ws     = (float*)d_ws;
  float* rv     = ws;                                   // B*R*MM
  float* rw     = rv + (size_t)B * R * MM;              // B*R*N
  float* wwb    = rw + (size_t)B * R * N;               // B*W*N
  float* Mem    = wwb + (size_t)B * W * N;              // B*N*MM
  float* instrs = Mem + (size_t)B * N * MM;             // B*NOUT
  __bf16* packWc = (__bf16*)(instrs + (size_t)B * NOUT);        // 1024 tiles * 512
  __bf16* packWk = packWc + (size_t)(KH / 32) * NT1 * 512;      // 960 tiles * 512

  k_init<<<256, 256, 0, stream>>>(rv, rw, wwb, Mem);
  k_pack<<<(KH / 32) * NT1 / 8, 256, 0, stream>>>(Wc, packWc, KH, CTRL, NT1);
  k_pack<<<(CTRL / 32) * NT2P / 8, 256, 0, stream>>>(Wk, packWk, CTRL, NOUT, NT2P);

  for (int t = 0; t < T; ++t) {
    k_controller<<<dim3(B / 16, CTRL / 64), 128, 0, stream>>>(x, packWc, bc, rv, out, t);
    k_instr<<<dim3(B / 16, 15), 128, 0, stream>>>(out, packWk, bk, instrs, t);
    k_update<<<B, 256, 0, stream>>>(instrs, Mem, rv, rw, wwb);
  }
}
